// ECRTM_73409581023500
// MI455X (gfx1250) — compile-verified, matrix operand order checked by reading, not run on hardware
//
#include <hip/hip_runtime.h>
#include <hip/hip_bf16.h>
#include <math.h>

#define DEV static __device__ __forceinline__

typedef __attribute__((ext_vector_type(16))) _Float16 v16h;
typedef __attribute__((ext_vector_type(8)))  float    v8f;

namespace ecrtm {

constexpr int Vn   = 5000;
constexpr int Bn   = 512;
constexpr int NTn  = 200;
constexpr int ENn  = 200;
constexpr int EMBn = 200;
constexpr int OD   = 40;
constexpr int KPAD = 256;            // EMB padded to multiple of 64 for pipelined K-loop
constexpr int VPAD = 5056;           // Vn padded to multiple of 64 (and 32) for tile loads
constexpr int NTPAD = 224;           // NTn padded to multiple of 32 for tile loads

union FragAB { v16h v; unsigned u[8]; uint4 q[2]; };
union FragC  { v8f  v; float    f[8]; };

// ---------- block reduction helpers (blockDim.x == 256) ----------
DEV float block_sum(float v, float* red) {
  const int tid = threadIdx.x;
  red[tid] = v; __syncthreads();
  for (int s = 128; s > 0; s >>= 1) {
    if (tid < s) red[tid] += red[tid + s];
    __syncthreads();
  }
  float r = red[0]; __syncthreads();
  return r;
}

DEV float block_max(float v, float* red) {
  const int tid = threadIdx.x;
  red[tid] = v; __syncthreads();
  for (int s = 128; s > 0; s >>= 1) {
    if (tid < s) red[tid] = fmaxf(red[tid], red[tid + s]);
    __syncthreads();
  }
  float r = red[0]; __syncthreads();
  return r;
}

// ---------- WMMA fragment load (unconditional; buffers are tile-padded) ----------
// 16-bit A-matrix 16x32 layout (ISA 7.12.2): lane = h*16+m holds row M=m;
// fragment VGPRs 0..3 = halves [h*8 .. h*8+7]  (16 contiguous bytes),
// fragment VGPRs 4..7 = halves [16+h*8 .. 16+h*8+7] (16 contiguous bytes, +32B).
// Same mapping serves B (32x16) when B is stored as rows of B^T (N,K) row-major.
DEV void load_frag(FragAB& f, const _Float16* base, int ld, int row0, int k0,
                   int lane) {
  const int h = lane >> 4, m = lane & 15;
  const uint4* q = (const uint4*)(base + (size_t)(row0 + m) * ld + k0 + h * 8);
  f.q[0] = q[0];   // bytes [0,16)  -> K pairs h*8 + {0,2,4,6}
  f.q[1] = q[2];   // bytes [32,48) -> K pairs 16 + h*8 + {0,2,4,6}
}

DEV void load_set(FragAB& a0, FragAB& a1, FragAB (&b)[4],
                  const _Float16* A, int lda, int row0,
                  const _Float16* Bm, int ldb, int col0, int k0, int lane) {
  load_frag(a0, A, lda, row0,      k0, lane);
  load_frag(a1, A, lda, row0 + 16, k0, lane);
#pragma unroll
  for (int t = 0; t < 4; ++t) load_frag(b[t], Bm, ldb, col0 + 16 * t, k0, lane);
}

DEV void do_wmma(FragC (&acc)[2][4], const FragAB& a0, const FragAB& a1,
                 const FragAB (&b)[4]) {
#pragma unroll
  for (int t = 0; t < 4; ++t) {
    acc[0][t].v = __builtin_amdgcn_wmma_f32_16x16x32_f16(
        false, a0.v, false, b[t].v, (short)0, acc[0][t].v, false, false);
    acc[1][t].v = __builtin_amdgcn_wmma_f32_16x16x32_f16(
        false, a1.v, false, b[t].v, (short)0, acc[1][t].v, false, false);
  }
}

// out[i*ldo + j] = sA[i] + sB[j] - 2 * (A_i . B_j);  A:(Mpad,K) B:(Npad,K) f16 row-major.
// One wave per 32x64 tile; double-buffered fragments so loads overlap WMMAs.
// __launch_bounds__(32, 1): single-wave workgroup, allow full VGPR budget (no spills);
// latency is hidden by the pipelined loads, not by occupancy.
// Requires K % 64 == 0 (buffers are zero-padded).
__global__ void __launch_bounds__(32, 1)
wmma_pdist(const _Float16* __restrict__ A, int lda, int M,
           const _Float16* __restrict__ Bm, int ldb, int N,
           int K, const float* __restrict__ sA,
           const float* __restrict__ sB,
           float* __restrict__ out, int ldo) {
  const int lane = threadIdx.x;
  const int row0 = blockIdx.x * 32;
  const int col0 = blockIdx.y * 64;
  FragC acc[2][4];
#pragma unroll
  for (int mi = 0; mi < 2; ++mi)
#pragma unroll
    for (int t = 0; t < 4; ++t)
#pragma unroll
      for (int r = 0; r < 8; ++r) acc[mi][t].f[r] = 0.f;

  FragAB a0A, a1A, bA[4], a0B, a1B, bB[4];
  load_set(a0A, a1A, bA, A, lda, row0, Bm, ldb, col0, 0, lane);
  for (int k0 = 0; k0 < K; k0 += 64) {
    load_set(a0B, a1B, bB, A, lda, row0, Bm, ldb, col0, k0 + 32, lane);
    do_wmma(acc, a0A, a1A, bA);
    if (k0 + 64 < K)
      load_set(a0A, a1A, bA, A, lda, row0, Bm, ldb, col0, k0 + 64, lane);
    do_wmma(acc, a0B, a1B, bB);
  }

  // D layout: lane = h*16+n, VGPR r -> (M = r + 8h, N = n)
  const int h = lane >> 4, nn = lane & 15;
#pragma unroll
  for (int mi = 0; mi < 2; ++mi) {
#pragma unroll
    for (int t = 0; t < 4; ++t) {
      const int j = col0 + 16 * t + nn;
#pragma unroll
      for (int r = 0; r < 8; ++r) {
        const int i = row0 + 16 * mi + r + 8 * h;
        if (i < M && j < N)
          out[(size_t)i * ldo + j] = sA[i] + sB[j] - 2.f * acc[mi][t].f[r];
      }
    }
  }
}

// ---------- prep kernels ----------
__global__ void init_ws(float* f0, float* u, float* scal) {
  const int idx = blockIdx.x * blockDim.x + threadIdx.x;
  if (idx < Vn)  f0[idx] = 0.f;
  if (idx < NTn) u[idx] = 1.f / (float)NTn;
  if (idx < 8)   scal[idx] = 0.f;
}

__global__ void make_X(const float* __restrict__ bow, _Float16* __restrict__ X,
                       int n, int npad, int d) {
  const int idx = blockIdx.x * blockDim.x + threadIdx.x;
  if (idx >= npad * d) return;
  const int i = idx / d, k = idx % d;
  X[idx] = (i < n) ? (_Float16)bow[(size_t)k * n + i] : (_Float16)0.f;  // X = bow.T
}

__global__ void make_pad16(const float* __restrict__ src, _Float16* __restrict__ dst,
                           int rows, int rowspad, int cols, int ldp) {
  const int idx = blockIdx.x * blockDim.x + threadIdx.x;
  if (idx >= rowspad * ldp) return;
  const int r = idx / ldp, c = idx % ldp;
  dst[idx] = (r < rows && c < cols) ? (_Float16)src[(size_t)r * cols + c]
                                    : (_Float16)0.f;
}

__global__ void colnorm(const float* __restrict__ bow, float* __restrict__ s,
                        int n, int d) {   // s_i = sum_k X[i,k]^2 = sum_k bow[k,i]^2
  const int i = blockIdx.x * blockDim.x + threadIdx.x;
  if (i >= n) return;
  float a = 0.f;
  for (int k = 0; k < d; ++k) { const float x = bow[(size_t)k * n + i]; a += x * x; }
  s[i] = a;
}

__global__ void rowsum(const float* __restrict__ bow, float* __restrict__ t,
                       int n, int d) {    // t_k = sum_i X[i,k]
  const int k = blockIdx.x * blockDim.x + threadIdx.x;
  if (k >= d) return;
  const float* r = bow + (size_t)k * n;
  float a = 0.f;
  for (int i = 0; i < n; ++i) a += r[i];
  t[k] = a;
}

__global__ void rownorm(const float* __restrict__ src, float* __restrict__ out,
                        int rows, int cols) {
  const int r = blockIdx.x * blockDim.x + threadIdx.x;
  if (r >= rows) return;
  float a = 0.f;
  const float* p = src + (size_t)r * cols;
  for (int c = 0; c < cols; ++c) a += p[c] * p[c];
  out[r] = a;
}

// mean(C) = (2n*sum_i s_i - 2*||sum_i x_i||^2)/n^2  -> eps, no 100MB reduction
__global__ void compute_eps(const float* __restrict__ s, const float* __restrict__ t,
                            float* __restrict__ scal, int n, int d) {
  __shared__ float red[256];
  const int tid = threadIdx.x;
  float a = 0.f;
  for (int i = tid; i < n; i += 256) a += s[i];
  const float S = block_sum(a, red);
  float b = 0.f;
  for (int k = tid; k < d; k += 256) b += t[k] * t[k];
  const float T = block_sum(b, red);
  if (tid == 0) {
    const float meanC = (2.f * (float)n * S - 2.f * T) / ((float)n * (float)n);
    scal[0] = meanC / logf(30.f) + 1e-6f;
  }
}

// ---------- big Sinkhorn: one block per row, online logsumexp over L2-resident C ----------
__global__ void sinkhorn_step(const float* __restrict__ C, const float* __restrict__ f,
                              float* __restrict__ fo, const float* __restrict__ scal,
                              int n, float loga) {
  const int i = blockIdx.x, tid = threadIdx.x;
  __shared__ float rm[256], rs[256];
  const float eps = scal[0], ieps = 1.f / eps;
  float m = -3.4e38f, ss = 0.f;
  const size_t rowo = (size_t)i * n;
  for (int j = tid; j < n; j += 256) {
    const float x = (f[j] - C[rowo + j]) * ieps;
    if (x > m) { ss = ss * __expf(m - x) + 1.f; m = x; }
    else       { ss += __expf(x - m); }
  }
  rm[tid] = m; rs[tid] = ss; __syncthreads();
  for (int s = 128; s > 0; s >>= 1) {
    if (tid < s) {
      const float m1 = rm[tid], m2 = rm[tid + s];
      const float M = fmaxf(m1, m2);
      rs[tid] = rs[tid] * __expf(m1 - M) + rs[tid + s] * __expf(m2 - M);
      rm[tid] = M;
    }
    __syncthreads();
  }
  if (tid == 0) {
    const float lse = rm[0] + __logf(rs[0]);
    fo[i] = 0.5f * (f[i] + eps * (loga - lse));
  }
}

// out[o*n + i] = sum_j exp((f_i + f_j - C_ij)/eps) * proj[j, o]
__global__ void p_matmul(const float* __restrict__ C, const float* __restrict__ f,
                         const float* __restrict__ scal, const float* __restrict__ proj,
                         float* __restrict__ out, int n) {
  const int i = blockIdx.x, tid = threadIdx.x;
  __shared__ float red[256];
  const float ieps = 1.f / scal[0];
  const float fi = f[i];
  float acc[OD];
#pragma unroll
  for (int o = 0; o < OD; ++o) acc[o] = 0.f;
  const size_t rowo = (size_t)i * n;
  for (int j = tid; j < n; j += 256) {
    const float p = __expf((fi + f[j] - C[rowo + j]) * ieps);
    const float4* pr = (const float4*)(proj + (size_t)j * OD);
#pragma unroll
    for (int q = 0; q < OD / 4; ++q) {
      const float4 v4 = pr[q];
      acc[4 * q + 0] += p * v4.x; acc[4 * q + 1] += p * v4.y;
      acc[4 * q + 2] += p * v4.z; acc[4 * q + 3] += p * v4.w;
    }
  }
  for (int o = 0; o < OD; ++o) {
    const float t = block_sum(acc[o], red);
    if (tid == 0) out[(size_t)o * n + i] = t;
  }
}

// ---------- encoder ----------
__global__ void linear_act(const float* __restrict__ A, const float* __restrict__ W,
                           const float* __restrict__ bias, float* __restrict__ out,
                           int M, int N, int K, int act) {
  const int idx = blockIdx.x * blockDim.x + threadIdx.x;
  if (idx >= M * N) return;
  const int m = idx / N, nn = idx % N;
  float s = bias[nn];
  const float* a = A + (size_t)m * K;
  const float* w = W + (size_t)nn * K;
  for (int k = 0; k < K; ++k) s += a[k] * w[k];
  if (act) s = (s > 20.f) ? s : log1pf(__expf(s));
  out[idx] = s;
}

__global__ void bnstats(const float* __restrict__ A, int M, int N,
                        float* __restrict__ mean, float* __restrict__ var) {
  const int j = blockIdx.x * blockDim.x + threadIdx.x;
  if (j >= N) return;
  float s = 0.f, s2 = 0.f;
  for (int m = 0; m < M; ++m) {
    const float x = A[(size_t)m * N + j];
    s += x; s2 += x * x;
  }
  const float mu = s / (float)M;
  mean[j] = mu;
  var[j]  = s2 / (float)M - mu * mu;
}

__global__ void bn_reparam(const float* __restrict__ mu_lin, const float* __restrict__ lv_lin,
                           const float* __restrict__ mmu, const float* __restrict__ vmu,
                           const float* __restrict__ mlv, const float* __restrict__ vlv,
                           const float* __restrict__ g_mu, const float* __restrict__ be_mu,
                           const float* __restrict__ g_lv, const float* __restrict__ be_lv,
                           const float* __restrict__ eps_noise,
                           float* __restrict__ muB, float* __restrict__ lvB,
                           float* __restrict__ zB) {
  const int idx = blockIdx.x * blockDim.x + threadIdx.x;
  if (idx >= OD * NTn) return;
  const int j = idx % NTn;
  const float mu = (mu_lin[idx] - mmu[j]) * rsqrtf(vmu[j] + 1e-5f) * g_mu[j] + be_mu[j];
  const float lv = (lv_lin[idx] - mlv[j]) * rsqrtf(vlv[j] + 1e-5f) * g_lv[j] + be_lv[j];
  muB[idx] = mu; lvB[idx] = lv;
  zB[idx] = mu + eps_noise[idx] * __expf(0.5f * lv);
}

__global__ void theta_softmax_kl(const float* __restrict__ zB, const float* __restrict__ muB,
                                 const float* __restrict__ lvB, float* __restrict__ theta,
                                 float* __restrict__ scal) {
  const int o = blockIdx.x, tid = threadIdx.x;
  __shared__ float red[256];
  float m = -3.4e38f;
  for (int j = tid; j < NTn; j += 256) m = fmaxf(m, zB[o * NTn + j]);
  m = block_max(m, red);
  float ss = 0.f;
  for (int j = tid; j < NTn; j += 256) ss += __expf(zB[o * NTn + j] - m);
  ss = block_sum(ss, red);
  const float inv = 1.f / ss;
  for (int j = tid; j < NTn; j += 256)
    theta[o * NTn + j] = __expf(zB[o * NTn + j] - m) * inv;
  const float var2 = 1.f - 1.f / (float)NTn;
  const float lv2 = logf(var2);
  float kl = 0.f;
  for (int j = tid; j < NTn; j += 256) {
    const float lv = lvB[o * NTn + j], muv = muB[o * NTn + j];
    kl += __expf(lv) / var2 + muv * muv / var2 + lv2 - lv;
  }
  kl = block_sum(kl, red);
  if (tid == 0) atomicAdd(&scal[1], 0.5f * (kl - (float)NTn) / (float)OD);
}

// ---------- small SNEkhorn (n=200) in one workgroup, theta staged in LDS ----------
__global__ void snek_small(const float* __restrict__ theta, const float* __restrict__ projK,
                           float* __restrict__ Cs, float* __restrict__ outA,
                           float* __restrict__ outB) {
  __shared__ float th[OD * NTn];
  __shared__ float sArr[NTn], fArr[NTn], fNew[NTn], red[256];
  __shared__ float epsS;
  const int tid = threadIdx.x;
  for (int idx = tid; idx < OD * NTn; idx += 256) th[idx] = theta[idx];
  __syncthreads();
  if (tid < NTn) {
    float s = 0.f;
    for (int k = 0; k < OD; ++k) { const float x = th[k * NTn + tid]; s += x * x; }
    sArr[tid] = s; fArr[tid] = 0.f;
  }
  __syncthreads();
  float lsum = 0.f;
  for (int idx = tid; idx < NTn * NTn; idx += 256) {
    const int i = idx / NTn, j = idx % NTn;
    float dot = 0.f;
    for (int k = 0; k < OD; ++k) dot += th[k * NTn + i] * th[k * NTn + j];
    const float c = sArr[i] + sArr[j] - 2.f * dot;
    Cs[idx] = c; lsum += c;
  }
  const float tot = block_sum(lsum, red);
  if (tid == 0) epsS = tot / (float)(NTn * NTn) / logf(30.f) + 1e-6f;
  __syncthreads();
  const float loga = -logf((float)NTn);
  for (int it = 0; it < 20; ++it) {
    if (tid < NTn) {
      const float ieps = 1.f / epsS;
      float m = -3.4e38f, ss = 0.f;
      for (int j = 0; j < NTn; ++j) {
        const float x = (fArr[j] - Cs[tid * NTn + j]) * ieps;
        if (x > m) { ss = ss * __expf(m - x) + 1.f; m = x; }
        else       { ss += __expf(x - m); }
      }
      const float lse = m + __logf(ss);
      fNew[tid] = 0.5f * (fArr[tid] + epsS * (loga - lse));
    }
    __syncthreads();
    if (tid < NTn) fArr[tid] = fNew[tid];
    __syncthreads();
  }
  if (tid < NTn) {
    const int i = tid;
    const float ieps = 1.f / epsS;
    float acc[OD];
#pragma unroll
    for (int o = 0; o < OD; ++o) acc[o] = 0.f;
    for (int j = 0; j < NTn; ++j) {
      const float p = __expf((fArr[i] + fArr[j] - Cs[i * NTn + j]) * ieps);
      const float* pr = projK + (size_t)j * OD;
#pragma unroll 8
      for (int o = 0; o < OD; ++o) acc[o] += p * pr[o];
    }
    for (int o = 0; o < OD; ++o) {
      outA[o * NTn + i] = acc[o];
      outB[o * NTn + i] = acc[o];
    }
  }
}

// ---------- decoder / losses ----------
__global__ void beta_softmax(const float* __restrict__ cost, float* __restrict__ beta,
                             int M, int N) {   // softmax over axis 0 of (-cost/0.2)
  const int j = blockIdx.x * blockDim.x + threadIdx.x;
  if (j >= N) return;
  float mx = -3.4e38f;
  for (int i = 0; i < M; ++i) mx = fmaxf(mx, -cost[(size_t)i * N + j] * 5.f);
  float ss = 0.f;
  for (int i = 0; i < M; ++i) ss += __expf(-cost[(size_t)i * N + j] * 5.f - mx);
  const float inv = 1.f / ss;
  for (int i = 0; i < M; ++i)
    beta[(size_t)i * N + j] = __expf(-cost[(size_t)i * N + j] * 5.f - mx) * inv;
}

__global__ void matmul_ab(const float* __restrict__ A, const float* __restrict__ Bm,
                          float* __restrict__ out, int M, int K, int N) {
  const int idx = blockIdx.x * blockDim.x + threadIdx.x;
  if (idx >= M * N) return;
  const int m = idx / N, j = idx % N;
  float s = 0.f;
  for (int k = 0; k < K; ++k) s += A[(size_t)m * K + k] * Bm[(size_t)k * N + j];
  out[idx] = s;
}

__global__ void recon_loss_k(float* __restrict__ rl, const float* __restrict__ mdec,
                             const float* __restrict__ vdec, const float* __restrict__ gdec,
                             const float* __restrict__ bdec, const float* __restrict__ bow_red,
                             float* __restrict__ scal) {
  const int o = blockIdx.x, tid = threadIdx.x;
  __shared__ float red[256];
  float m = -3.4e38f;
  for (int j = tid; j < Vn; j += 256) {
    const float y = (rl[(size_t)o * Vn + j] - mdec[j]) * rsqrtf(vdec[j] + 1e-5f) * gdec[j] + bdec[j];
    rl[(size_t)o * Vn + j] = y;
    m = fmaxf(m, y);
  }
  m = block_max(m, red);
  float ss = 0.f;
  for (int j = tid; j < Vn; j += 256) ss += __expf(rl[(size_t)o * Vn + j] - m);
  ss = block_sum(ss, red);
  const float lognorm = m + __logf(ss);
  float loss = 0.f;
  for (int j = tid; j < Vn; j += 256)
    loss -= bow_red[(size_t)o * Vn + j] * (rl[(size_t)o * Vn + j] - lognorm);
  loss = block_sum(loss, red);
  if (tid == 0) atomicAdd(&scal[2], loss / (float)OD);
}

// ---------- ECR ----------
__global__ void km_kernel(const float* __restrict__ cost, float* __restrict__ Km, int sz) {
  const int idx = blockIdx.x * blockDim.x + threadIdx.x;
  if (idx >= sz) return;
  Km[idx] = __expf(-20.f * cost[idx]);
}

__global__ void ecr_v(const float* __restrict__ Km, const float* __restrict__ u,
                      float* __restrict__ v, int M, int N) {
  const int j = blockIdx.x * blockDim.x + threadIdx.x;
  if (j >= N) return;
  float acc = 0.f;
  for (int i = 0; i < M; ++i) acc += Km[(size_t)i * N + j] * u[i];
  v[j] = (1.f / (float)N) / (acc + 1e-16f);
}

__global__ void ecr_u(const float* __restrict__ Km, const float* __restrict__ v,
                      float* __restrict__ u, int M, int N) {
  const int i = blockIdx.x, tid = threadIdx.x;
  __shared__ float red[256];
  float acc = 0.f;
  const float* row = Km + (size_t)i * N;
  for (int j = tid; j < N; j += 256) acc += row[j] * v[j];
  acc = block_sum(acc, red);
  if (tid == 0) u[i] = (1.f / (float)M) / (acc + 1e-16f);
}

__global__ void ecr_loss_k(const float* __restrict__ Km, const float* __restrict__ cost,
                           const float* __restrict__ u, const float* __restrict__ v,
                           float* __restrict__ scal, int M, int N) {
  __shared__ float red[256];
  const size_t total = (size_t)M * N;
  float acc = 0.f;
  for (size_t idx = (size_t)blockIdx.x * blockDim.x + threadIdx.x; idx < total;
       idx += (size_t)gridDim.x * blockDim.x) {
    const int i = (int)(idx / N), j = (int)(idx % N);
    acc += u[i] * Km[idx] * v[j] * cost[idx];
  }
  acc = block_sum(acc, red);
  if (threadIdx.x == 0) atomicAdd(&scal[3], acc);
}

__global__ void finalize(const float* __restrict__ scal, float* __restrict__ out) {
  const float tm  = scal[2] + scal[1];
  const float ecr = 250.f * scal[3];
  out[0] = tm + ecr;
  out[1] = tm;
  out[2] = ecr;
}

}  // namespace ecrtm

extern "C" void kernel_launch(void* const* d_in, const int* in_sizes, int n_in,
                              void* d_out, int out_size, void* d_ws, size_t ws_size,
                              hipStream_t stream) {
  using namespace ecrtm;
  (void)in_sizes; (void)n_in; (void)out_size; (void)ws_size;

  const float* bow       = (const float*)d_in[0];
  const float* eps_noise = (const float*)d_in[1];
  const float* W1  = (const float*)d_in[2];  const float* b1  = (const float*)d_in[3];
  const float* W2  = (const float*)d_in[4];  const float* b2  = (const float*)d_in[5];
  const float* Wmu = (const float*)d_in[6];  const float* bmu = (const float*)d_in[7];
  const float* Wlv = (const float*)d_in[8];  const float* blv = (const float*)d_in[9];
  const float* g_mu  = (const float*)d_in[10]; const float* be_mu  = (const float*)d_in[11];
  const float* g_lv  = (const float*)d_in[12]; const float* be_lv  = (const float*)d_in[13];
  const float* g_dec = (const float*)d_in[14]; const float* be_dec = (const float*)d_in[15];
  const float* word_emb  = (const float*)d_in[16];
  const float* topic_emb = (const float*)d_in[17];
  const float* proj_V    = (const float*)d_in[18];
  const float* proj_K    = (const float*)d_in[19];
  float* out = (float*)d_out;

  char* base = (char*)d_ws;
  size_t off = 0;
  auto alloc = [&](size_t bytes) -> void* {
    void* p = base + off;
    off = (off + bytes + 255) & ~(size_t)255;
    return p;
  };

  float*    Cbig = (float*)alloc((size_t)Vn * Vn * 4);      // 100 MB, L2-resident
  _Float16* X16  = (_Float16*)alloc((size_t)VPAD * Bn * 2); // zero-padded rows
  _Float16* w16  = (_Float16*)alloc((size_t)VPAD * KPAD * 2);
  _Float16* t16  = (_Float16*)alloc((size_t)NTPAD * KPAD * 2);
  float* sbig = (float*)alloc(Vn * 4);
  float* tvec = (float*)alloc(Bn * 4);
  float* f0 = (float*)alloc(Vn * 4);
  float* f1 = (float*)alloc(Vn * 4);
  float* bow_red = (float*)alloc((size_t)OD * Vn * 4);
  float* e1 = (float*)alloc(OD * ENn * 4);
  float* e2 = (float*)alloc(OD * ENn * 4);
  float* mu_lin = (float*)alloc(OD * NTn * 4);
  float* lv_lin = (float*)alloc(OD * NTn * 4);
  float* muB = (float*)alloc(OD * NTn * 4);
  float* lvB = (float*)alloc(OD * NTn * 4);
  float* zB  = (float*)alloc(OD * NTn * 4);
  float* theta = (float*)alloc(OD * NTn * 4);
  float* mmu = (float*)alloc(NTn * 4); float* vmu = (float*)alloc(NTn * 4);
  float* mlv = (float*)alloc(NTn * 4); float* vlv = (float*)alloc(NTn * 4);
  float* Csmall = (float*)alloc(NTn * NTn * 4);
  float* theta_red = (float*)alloc(OD * NTn * 4);
  float* tsq = (float*)alloc(NTn * 4); float* wsq = (float*)alloc(Vn * 4);
  float* cost = (float*)alloc((size_t)NTn * Vn * 4);
  float* beta = (float*)alloc((size_t)NTn * Vn * 4);
  float* Km   = (float*)alloc((size_t)NTn * Vn * 4);
  float* rl   = (float*)alloc((size_t)OD * Vn * 4);
  float* mdec = (float*)alloc(Vn * 4); float* vdec = (float*)alloc(Vn * 4);
  float* uvec = (float*)alloc(NTn * 4); float* vvec = (float*)alloc(Vn * 4);
  float* scal = (float*)alloc(64 * 4);  // [0]=eps, [1]=KL, [2]=recon, [3]=ECR-sum

  auto cdiv = [](int a, int b) { return (a + b - 1) / b; };

  // ---- init + prep ----
  init_ws<<<cdiv(Vn, 256), 256, 0, stream>>>(f0, uvec, scal);
  make_X<<<cdiv(VPAD * Bn, 256), 256, 0, stream>>>(bow, X16, Vn, VPAD, Bn);
  colnorm<<<cdiv(Vn, 256), 256, 0, stream>>>(bow, sbig, Vn, Bn);
  rowsum<<<cdiv(Bn, 256), 256, 0, stream>>>(bow, tvec, Vn, Bn);
  compute_eps<<<1, 256, 0, stream>>>(sbig, tvec, scal, Vn, Bn);

  // ---- big pairwise distances via f16 WMMA (f32 accumulate), 32x64 tiles ----
  wmma_pdist<<<dim3(cdiv(Vn, 32), cdiv(Vn, 64)), 32, 0, stream>>>(
      X16, Bn, Vn, X16, Bn, Vn, Bn, sbig, sbig, Cbig, Vn);

  // ---- 20 symmetric Sinkhorn sweeps over L2-resident C ----
  {
    const float loga = -logf((float)Vn);
    float* fa = f0; float* fb = f1;
    for (int it = 0; it < 20; ++it) {
      sinkhorn_step<<<Vn, 256, 0, stream>>>(Cbig, fa, fb, scal, Vn, loga);
      float* t_ = fa; fa = fb; fb = t_;
    }
    p_matmul<<<Vn, 256, 0, stream>>>(Cbig, fa, scal, proj_V, bow_red, Vn);
  }

  // ---- encoder ----
  linear_act<<<cdiv(OD * ENn, 256), 256, 0, stream>>>(bow_red, W1, b1, e1, OD, ENn, Vn, 1);
  linear_act<<<cdiv(OD * ENn, 256), 256, 0, stream>>>(e1, W2, b2, e2, OD, ENn, ENn, 1);
  linear_act<<<cdiv(OD * NTn, 256), 256, 0, stream>>>(e2, Wmu, bmu, mu_lin, OD, NTn, ENn, 0);
  linear_act<<<cdiv(OD * NTn, 256), 256, 0, stream>>>(e2, Wlv, blv, lv_lin, OD, NTn, ENn, 0);
  bnstats<<<cdiv(NTn, 256), 256, 0, stream>>>(mu_lin, OD, NTn, mmu, vmu);
  bnstats<<<cdiv(NTn, 256), 256, 0, stream>>>(lv_lin, OD, NTn, mlv, vlv);
  bn_reparam<<<cdiv(OD * NTn, 256), 256, 0, stream>>>(
      mu_lin, lv_lin, mmu, vmu, mlv, vlv, g_mu, be_mu, g_lv, be_lv,
      eps_noise, muB, lvB, zB);
  theta_softmax_kl<<<OD, 256, 0, stream>>>(zB, muB, lvB, theta, scal);

  // ---- small SNEkhorn on theta.T; theta_reduced is also an output ----
  snek_small<<<1, 256, 0, stream>>>(theta, proj_K, Csmall, theta_red, out + 3);

  // ---- cost = pairwise(topic, word) via WMMA; beta; decode ----
  make_pad16<<<cdiv(VPAD * KPAD, 256), 256, 0, stream>>>(word_emb, w16, Vn, VPAD, EMBn, KPAD);
  make_pad16<<<cdiv(NTPAD * KPAD, 256), 256, 0, stream>>>(topic_emb, t16, NTn, NTPAD, EMBn, KPAD);
  rownorm<<<cdiv(Vn, 256), 256, 0, stream>>>(word_emb, wsq, Vn, EMBn);
  rownorm<<<cdiv(NTn, 256), 256, 0, stream>>>(topic_emb, tsq, NTn, EMBn);
  wmma_pdist<<<dim3(cdiv(NTn, 32), cdiv(Vn, 64)), 32, 0, stream>>>(
      t16, KPAD, NTn, w16, KPAD, Vn, KPAD, tsq, wsq, cost, Vn);
  beta_softmax<<<cdiv(Vn, 256), 256, 0, stream>>>(cost, beta, NTn, Vn);
  matmul_ab<<<cdiv(OD * Vn, 256), 256, 0, stream>>>(theta_red, beta, rl, OD, NTn, Vn);
  bnstats<<<cdiv(Vn, 256), 256, 0, stream>>>(rl, OD, Vn, mdec, vdec);
  recon_loss_k<<<OD, 256, 0, stream>>>(rl, mdec, vdec, g_dec, be_dec, bow_red, scal);

  // ---- ECR entropic OT (100 fixed iterations, Km L2-resident) ----
  km_kernel<<<cdiv(NTn * Vn, 256), 256, 0, stream>>>(cost, Km, NTn * Vn);
  for (int it = 0; it < 100; ++it) {
    ecr_v<<<cdiv(Vn, 256), 256, 0, stream>>>(Km, uvec, vvec, NTn, Vn);
    ecr_u<<<NTn, 256, 0, stream>>>(Km, vvec, uvec, NTn, Vn);
  }
  ecr_v<<<cdiv(Vn, 256), 256, 0, stream>>>(Km, uvec, vvec, NTn, Vn);
  ecr_loss_k<<<512, 256, 0, stream>>>(Km, cost, uvec, vvec, scal, NTn, Vn);

  finalize<<<1, 1, 0, stream>>>(scal, out);
}
